// AttnBlock_16114717295006
// MI455X (gfx1250) — matrix operand order repaired and checked by reading, not compiled, not run on hardware
//
#include <hip/hip_runtime.h>
#include <hip/hip_bf16.h>
#include <math.h>

// ---------------------------------------------------------------------------
// AttnBlock for MI455X (gfx1250, wave32, WMMA).
// Shapes fixed by the reference: B=4, C=512, H=W=64, HW=4096.
// All heavy math routed through v_wmma_f32_16x16x32_f16 (f16 in, f32 acc).
// Workspace layout (f16 elements), ~94.4 MB total:
//   wq_h[C*C] | wk_h[9*C*C] | wv_h[9*C*C] | wp_h[C*C] |
//   h_t[B*HW*C]  ([b][p][c])   normalized input
//   q_t[B*HW*C]  ([b][p][c])   scaled by 1/sqrt(C)
//   k_t[B*HW*C]  ([b][p][c])
//   v_t[B*HW*C]  ([b][c][p])
//   ao_t[B*HW*C] ([b][p][c])   attention output
// ---------------------------------------------------------------------------

typedef _Float16 h8   __attribute__((ext_vector_type(8)));
typedef _Float16 v16h __attribute__((ext_vector_type(16)));
typedef float    v8f  __attribute__((ext_vector_type(8)));

#define CC   512
#define HH   64
#define WW   64
#define HWHW 4096
#define BB   4

// ---- WMMA fragment loaders (layouts per CDNA5 ISA 7.12.2, wave32) ---------
// A 16x32 f16: lane L: M = L&15, holds K = klo..klo+7 and 16+klo..16+klo+7,
//              klo = (L>>4)*8.
__device__ __forceinline__ v16h load_a16(const _Float16* base, int ldk) {
    const int lane = threadIdx.x & 31;
    const int row  = lane & 15;
    const int klo  = (lane >> 4) << 3;
    const h8 lo = *reinterpret_cast<const h8*>(base + (size_t)row * ldk + klo);
    const h8 hi = *reinterpret_cast<const h8*>(base + (size_t)row * ldk + 16 + klo);
    return __builtin_shufflevector(lo, hi, 0,1,2,3,4,5,6,7,8,9,10,11,12,13,14,15);
}
// B 32x16 f16 stored as [N][K]: lane L: N = L&15, holds K = kb..kb+7 and
//              16+kb..16+kb+7, kb = (L>>4)*8 (same K order as A).
__device__ __forceinline__ v16h load_b16(const _Float16* base, int ldk) {
    const int lane = threadIdx.x & 31;
    const int row  = lane & 15;
    const int kb   = (lane >> 4) << 3;
    const h8 lo = *reinterpret_cast<const h8*>(base + (size_t)row * ldk + kb);
    const h8 hi = *reinterpret_cast<const h8*>(base + (size_t)row * ldk + 16 + kb);
    return __builtin_shufflevector(lo, hi, 0,1,2,3,4,5,6,7,8,9,10,11,12,13,14,15);
}
__device__ __forceinline__ v8f wmma16(v16h a, v16h b, v8f c) {
    return __builtin_amdgcn_wmma_f32_16x16x32_f16(false, a, false, b, (short)0, c,
                                                  false, false);
}
// C/D 16x16 f32: VGPR r, lane L: M = r + 8*(L>>4), N = L&15.

// ---------------------------------------------------------------------------
// Weight prep: f32 -> f16; 3x3 weights reordered OIHW -> [tap][O][I].
__global__ __launch_bounds__(256) void prep_weights_kernel(
        const float* __restrict__ wq, const float* __restrict__ wk,
        const float* __restrict__ wv, const float* __restrict__ wp,
        _Float16* __restrict__ wq_h, _Float16* __restrict__ wk_h,
        _Float16* __restrict__ wv_h, _Float16* __restrict__ wp_h) {
    const int i = blockIdx.x * blockDim.x + threadIdx.x;
    const int n9 = 9 * CC * CC;
    if (i < n9) {
        const int o   = i / (CC * 9);
        const int rem = i - o * (CC * 9);
        const int ci  = rem / 9;
        const int t   = rem - ci * 9;
        const int dst = t * CC * CC + o * CC + ci;
        wk_h[dst] = (_Float16)wk[i];
        wv_h[dst] = (_Float16)wv[i];
    }
    if (i < CC * CC) {
        wq_h[i] = (_Float16)wq[i];
        wp_h[i] = (_Float16)wp[i];
    }
}

// ---------------------------------------------------------------------------
// Channels-first LayerNorm; writes h_t f16 in [b][p][c] (K-contiguous).
__global__ __launch_bounds__(256) void layernorm_kernel(
        const float* __restrict__ x, const float* __restrict__ ln_w,
        const float* __restrict__ ln_b, _Float16* __restrict__ h_t) {
    const int pix = blockIdx.x * blockDim.x + threadIdx.x;   // [0, B*HW)
    const int b = pix >> 12;           // / HW
    const int p = pix & (HWHW - 1);
    const float* xb = x + (size_t)b * CC * HWHW + p;
    float s = 0.f, s2 = 0.f;
    for (int c = 0; c < CC; ++c) {
        const float v = xb[(size_t)c * HWHW];
        s += v; s2 += v * v;
    }
    const float mean = s * (1.0f / CC);
    const float var  = s2 * (1.0f / CC) - mean * mean;
    const float rstd = rsqrtf(var + 1e-6f);
    _Float16* hb = h_t + ((size_t)b * HWHW + p) * CC;
    for (int c = 0; c < CC; ++c) {
        const float v = (xb[(size_t)c * HWHW] - mean) * rstd * ln_w[c] + ln_b[c];
        hb[c] = (_Float16)v;
    }
}

// ---------------------------------------------------------------------------
// Generic NT-GEMM over WMMA: out[cout][pix] = sum_{tap,cin} A[tap][cout][cin] *
//   Bsrc[b][pix_shift(tap)][cin], with 3x3 border masking when NTAPS==9.
// MODE 0: f16 store [b][p][c] (transposed)  (q, k)
// MODE 1: f16 store [b][c][p]               (v)
// MODE 2: f32 store x + bias + val to d_out (proj)
template <int NTAPS, int MODE>
__global__ __launch_bounds__(256) void gemm_nt_kernel(
        const _Float16* __restrict__ A, const _Float16* __restrict__ Bsrc,
        _Float16* __restrict__ outh, float* __restrict__ outf,
        const float* __restrict__ xres, const float* __restrict__ bias,
        float scale) {
    constexpr int LDK = 40;            // padded LDS row (16B-aligned fragments)
    __shared__ __align__(16) _Float16 As[64 * LDK];
    __shared__ __align__(16) _Float16 Bs[64 * LDK];

    const int pb = blockIdx.x * 64;    // pixel tile base
    const int cb = blockIdx.y * 64;    // cout tile base
    const int b  = blockIdx.z;
    const int tid  = threadIdx.x;
    const int lane = tid & 31;
    const int wave = tid >> 5;
    const int wr = wave & 3;           // cout 16-row group
    const int wc = wave >> 2;          // pixel 32-col group

    // staging coords: 8 halves per thread
    const int srow = (tid * 8) >> 5;
    const int scol = (tid * 8) & 31;

    v8f acc0 = {}, acc1 = {};

    for (int tap = 0; tap < NTAPS; ++tap) {
        const int dy = (NTAPS == 9) ? (tap / 3 - 1) : 0;
        const int dx = (NTAPS == 9) ? (tap % 3 - 1) : 0;
        for (int kc = 0; kc < CC / 32; ++kc) {
            __syncthreads();
            // A chunk: [64 cout][32 cin]
            {
                const h8 v = *reinterpret_cast<const h8*>(
                    A + (size_t)tap * CC * CC + (size_t)(cb + srow) * CC + kc * 32 + scol);
                *reinterpret_cast<h8*>(&As[srow * LDK + scol]) = v;
            }
            // B chunk: [64 pix][32 cin], shifted + masked for 3x3
            {
                const int p = pb + srow;
                const int y = p >> 6, xq = p & 63;
                const int yy = y + dy, xx = xq + dx;
                h8 v = {};
                if ((unsigned)yy < (unsigned)HH && (unsigned)xx < (unsigned)WW) {
                    const int sp = yy * WW + xx;
                    v = *reinterpret_cast<const h8*>(
                        Bsrc + ((size_t)b * HWHW + sp) * CC + kc * 32 + scol);
                }
                *reinterpret_cast<h8*>(&Bs[srow * LDK + scol]) = v;
            }
            __syncthreads();
            const v16h a  = load_a16(&As[wr * 16 * LDK], LDK);
            const v16h b0 = load_b16(&Bs[(wc * 32) * LDK], LDK);
            const v16h b1 = load_b16(&Bs[(wc * 32 + 16) * LDK], LDK);
            acc0 = wmma16(a, b0, acc0);
            acc1 = wmma16(a, b1, acc1);
        }
    }

    // epilogue
#pragma unroll
    for (int f = 0; f < 2; ++f) {
        const v8f acc = f ? acc1 : acc0;
#pragma unroll
        for (int r = 0; r < 8; ++r) {
            const int cm = cb + wr * 16 + r + 8 * (lane >> 4);
            const int pn = pb + wc * 32 + f * 16 + (lane & 15);
            const float val = acc[r] * scale;
            if (MODE == 0) {
                outh[((size_t)b * HWHW + pn) * CC + cm] = (_Float16)val;
            } else if (MODE == 1) {
                outh[((size_t)b * CC + cm) * HWHW + pn] = (_Float16)val;
            } else {
                const size_t o = (size_t)b * CC * HWHW + (size_t)cm * HWHW + pn;
                outf[o] = xres[o] + bias[cm] + val;
            }
        }
    }
}

// ---------------------------------------------------------------------------
// Flash-style attention. Block = 8 waves = 2 query groups of 4 waves; each
// group owns 16 query rows, each wave of a group owns 128 output channels
// (8 x 16x16 f32 accumulators -> 64 VGPRs, no spills). The sub-0 wave of a
// group computes S = Q^T K and the running base-2 softmax; P / corrections
// are shared through LDS.
__global__ __launch_bounds__(256, 1) void attn_kernel(
        const _Float16* __restrict__ q_t, const _Float16* __restrict__ k_t,
        const _Float16* __restrict__ v_t, _Float16* __restrict__ ao) {
    __shared__ __align__(16) _Float16 Ps[2 * 16 * 64];
    __shared__ float corr_s[2 * 16];
    __shared__ float invl_s[2 * 16];

    const int mt   = blockIdx.x & (HWHW / 32 - 1);   // 128 m-tiles per batch
    const int b    = blockIdx.x / (HWHW / 32);
    const int tid  = threadIdx.x;
    const int lane = tid & 31;
    const int wave = tid >> 5;
    const int grp  = wave >> 2;        // query group (16 rows each)
    const int sub  = wave & 3;         // 128-channel slice within group
    const int mbase = mt * 32 + grp * 16;
    const int hio   = lane >> 4;       // half-of-wave row offset selector

    const _Float16* qb = q_t + ((size_t)b * HWHW + mbase) * CC;
    const _Float16* kb = k_t + (size_t)b * HWHW * CC;
    const _Float16* vb = v_t + (size_t)b * CC * HWHW + (size_t)sub * 128 * HWHW;

    v8f o_acc[8];
#pragma unroll
    for (int i = 0; i < 8; ++i) o_acc[i] = (v8f){};
    float mrun[8], lrun[8];
#pragma unroll
    for (int r = 0; r < 8; ++r) { mrun[r] = -1e30f; lrun[r] = 0.f; }
    const float LOG2E = 1.44269504088896f;

    for (int nb = 0; nb < HWHW; nb += 64) {
        __syncthreads();               // Ps consumers of prev tile done
        if (sub == 0) {
            v8f s[4] = {(v8f){}, (v8f){}, (v8f){}, (v8f){}};
            for (int kc = 0; kc < CC / 32; ++kc) {
                const v16h aq = load_a16(qb + kc * 32, CC);
#pragma unroll
                for (int j = 0; j < 4; ++j) {
                    const v16h bk = load_b16(kb + (size_t)(nb + j * 16) * CC + kc * 32, CC);
                    s[j] = wmma16(aq, bk, s[j]);
                }
            }
            // running softmax (base-2 domain; 1/sqrt(C) already folded into q)
#pragma unroll
            for (int r = 0; r < 8; ++r) {
                float mx = -1e30f;
#pragma unroll
                for (int j = 0; j < 4; ++j) mx = fmaxf(mx, s[j][r] * LOG2E);
                for (int off = 1; off < 16; off <<= 1)
                    mx = fmaxf(mx, __shfl_xor(mx, off, 16));
                const float nm = fmaxf(mrun[r], mx);
                const float cr = exp2f(mrun[r] - nm);
                const int   m  = r + 8 * hio;
                float rs = 0.f;
#pragma unroll
                for (int j = 0; j < 4; ++j) {
                    const float p = exp2f(s[j][r] * LOG2E - nm);
                    Ps[grp * 1024 + m * 64 + (lane & 15) + 16 * j] = (_Float16)p;
                    rs += p;
                }
                for (int off = 1; off < 16; off <<= 1)
                    rs += __shfl_xor(rs, off, 16);
                lrun[r] = lrun[r] * cr + rs;
                mrun[r] = nm;
                if ((lane & 15) == 0) corr_s[grp * 16 + m] = cr;
            }
        }
        __syncthreads();
        // rescale O by correction
#pragma unroll
        for (int r = 0; r < 8; ++r) {
            const float cr = corr_s[grp * 16 + r + 8 * hio];
#pragma unroll
            for (int cf = 0; cf < 8; ++cf) o_acc[cf][r] *= cr;
        }
        // O += P x V
        const v16h p0 = load_a16(&Ps[grp * 1024], 64);
        const v16h p1 = load_a16(&Ps[grp * 1024 + 32], 64);
#pragma unroll
        for (int cf = 0; cf < 8; ++cf) {
            const _Float16* vrow = vb + (size_t)(cf * 16) * HWHW + nb;
            const v16h v0 = load_b16(vrow, HWHW);
            const v16h v1 = load_b16(vrow + 32, HWHW);
            o_acc[cf] = wmma16(p0, v0, o_acc[cf]);
            o_acc[cf] = wmma16(p1, v1, o_acc[cf]);
        }
    }

    __syncthreads();
    if (sub == 0 && (lane & 15) == 0) {
#pragma unroll
        for (int r = 0; r < 8; ++r)
            invl_s[grp * 16 + r + 8 * hio] = 1.0f / lrun[r];
    }
    __syncthreads();
#pragma unroll
    for (int r = 0; r < 8; ++r) {
        const float il = invl_s[grp * 16 + r + 8 * hio];
        const int m = mbase + r + 8 * hio;
#pragma unroll
        for (int cf = 0; cf < 8; ++cf) {
            const int c = sub * 128 + cf * 16 + (lane & 15);
            ao[((size_t)b * HWHW + m) * CC + c] = (_Float16)(o_acc[cf][r] * il);
        }
    }
}

// ---------------------------------------------------------------------------
extern "C" void kernel_launch(void* const* d_in, const int* in_sizes, int n_in,
                              void* d_out, int out_size, void* d_ws, size_t ws_size,
                              hipStream_t stream) {
    const float* x    = (const float*)d_in[0];
    const float* ln_w = (const float*)d_in[1];
    const float* ln_b = (const float*)d_in[2];
    const float* wq   = (const float*)d_in[3];
    const float* wk   = (const float*)d_in[4];
    const float* wv   = (const float*)d_in[5];
    const float* wp   = (const float*)d_in[6];
    const float* bp   = (const float*)d_in[7];
    float* out = (float*)d_out;

    const size_t SZ_W1 = (size_t)CC * CC;            // 262144
    const size_t SZ_W9 = 9 * SZ_W1;
    const size_t SZ_T  = (size_t)BB * HWHW * CC;     // 8388608
    _Float16* ws   = (_Float16*)d_ws;                // needs ~94.4 MB
    _Float16* wq_h = ws;
    _Float16* wk_h = wq_h + SZ_W1;
    _Float16* wv_h = wk_h + SZ_W9;
    _Float16* wp_h = wv_h + SZ_W9;
    _Float16* h_t  = wp_h + SZ_W1;
    _Float16* q_t  = h_t + SZ_T;
    _Float16* k_t  = q_t + SZ_T;
    _Float16* v_t  = k_t + SZ_T;
    _Float16* ao_t = v_t + SZ_T;

    const float qscale = 0.044194173824159216f;      // 1/sqrt(512)

    prep_weights_kernel<<<(int)((SZ_W9 + 255) / 256), 256, 0, stream>>>(
        wq, wk, wv, wp, wq_h, wk_h, wv_h, wp_h);

    layernorm_kernel<<<(BB * HWHW) / 256, 256, 0, stream>>>(x, ln_w, ln_b, h_t);

    dim3 ggrid(HWHW / 64, CC / 64, BB);
    gemm_nt_kernel<1, 0><<<ggrid, 256, 0, stream>>>(wq_h, h_t, q_t, nullptr,
                                                    nullptr, nullptr, qscale);
    gemm_nt_kernel<9, 0><<<ggrid, 256, 0, stream>>>(wk_h, h_t, k_t, nullptr,
                                                    nullptr, nullptr, 1.0f);
    gemm_nt_kernel<9, 1><<<ggrid, 256, 0, stream>>>(wv_h, h_t, v_t, nullptr,
                                                    nullptr, nullptr, 1.0f);

    attn_kernel<<<BB * (HWHW / 32), 256, 0, stream>>>(q_t, k_t, v_t, ao_t);

    gemm_nt_kernel<1, 2><<<ggrid, 256, 0, stream>>>(wp_h, ao_t, nullptr, out,
                                                    x, bp, 1.0f);
}
